// SAGEResInception_5282809775003
// MI455X (gfx1250) — matrix operand order repaired and checked by reading, not compiled, not run on hardware
//
#include <hip/hip_runtime.h>
#include <hip/hip_bf16.h>

// ---------------------------------------------------------------------------
// SAGE + Res + Inception + MLP pipeline for gfx1250 (MI455X).
// Dense math via v_wmma_f32_16x16x32_bf16 (wave32 WMMA), fp32 accumulate.
// Block tile 128x128 (8 waves, 32x64 per wave = 8 accumulators), weights
// pre-transposed to bf16 [n][k]; double-buffered LDS, register-staged global
// loads (segment switch keeps A loads in the global address space).
// ---------------------------------------------------------------------------

#define HDIM 512
#define ODIM 256
#define BN_EPS 1e-5f

typedef __bf16 bf16;
typedef __attribute__((ext_vector_type(16))) __bf16 v16bf;
typedef __attribute__((ext_vector_type(8)))  __bf16 v8bf;
typedef __attribute__((ext_vector_type(8)))  float  v8f;

// ------------------------------ utility kernels ----------------------------

__global__ void zero_f32(float* p, long n) {
    long i = (long)blockIdx.x * blockDim.x + threadIdx.x;
    long stride = (long)gridDim.x * blockDim.x;
    for (; i < n; i += stride) p[i] = 0.0f;
}

// dst[n*Ktot + koff + k] = (bf16)src[k*N + n]  -- transpose+convert weights
__global__ void transpose_cvt_bf16(const float* __restrict__ src, bf16* __restrict__ dst,
                                   int Ksrc, int N, int Ktot, int koff) {
    int i = blockIdx.x * blockDim.x + threadIdx.x;
    if (i >= Ksrc * N) return;
    int k = i / N, n = i - k * N;
    dst[(size_t)n * Ktot + koff + k] = (bf16)src[i];
}

// ------------------------ segment mean (edge scatter) ----------------------
// One block per edge; 512-wide feature rows; fp32 atomics (HBM-bound stage).

__global__ void scatter_mean_accum(const float* __restrict__ x,
                                   const int* __restrict__ src,
                                   const int* __restrict__ dst,
                                   int E, float* ssum, float* cnt) {
    int e = blockIdx.x;
    if (e >= E) return;
    int s = src[e], d = dst[e];
    const float* xs = x + (size_t)s * HDIM;
    float* out = ssum + (size_t)d * HDIM;
    for (int f = threadIdx.x; f < HDIM; f += blockDim.x)
        atomicAdd(&out[f], xs[f]);
    if (threadIdx.x == 0) atomicAdd(&cnt[d], 1.0f);
}

__global__ void mean_div(float* ssum, const float* __restrict__ cnt, int ndst) {
    int i = blockIdx.x * blockDim.x + threadIdx.x;
    if (i < ndst * HDIM) {
        float c = cnt[i >> 9];                 // row = i / 512
        ssum[i] = ssum[i] / fmaxf(c, 1.0f);
    }
}

// ------------------------------- WMMA GEMM ---------------------------------
// C[M x N] = concat_k(A0|A1|A2|A3)[M x K] @ W[K x N] (+ bias)
//  * A segments: fp32, each 512 columns wide, row stride 512
//  * Wt: bf16, TRANSPOSED [n][k], row stride K  (fragment-ready layout)
//  * block: 256 threads (8 wave32), 128x128 C tile, K step 32
//  * per wave: 32x64 C strip -> 2x4 v8f accumulators, 8 WMMA per K-step
//  * double-buffered LDS, one barrier per K-step, loads for step i+1 issued
//    while step i computes; per-segment switch keeps A loads global_load.

__launch_bounds__(256)
__global__ void wmma_gemm(const float* A0, const float* A1,
                          const float* A2, const float* A3,
                          int K, const bf16* __restrict__ Wt, int N, int M,
                          const float* __restrict__ bias, float* __restrict__ C) {
    __shared__ __align__(16) bf16 Asm[2][128 * 32];   // [m][k]
    __shared__ __align__(16) bf16 Bsm[2][128 * 32];   // [n][k]

    const int tid  = threadIdx.x;
    const int lane = tid & 31;
    const int wid  = tid >> 5;
    const int wm   = wid & 3;           // wave row  (4 x 32 = 128)
    const int wn   = wid >> 2;          // wave col  (2 x 64 = 128)
    const int m_blk = blockIdx.y * 128;
    const int n_blk = blockIdx.x * 128;
    const int hi   = lane >> 4;
    const int l15  = lane & 15;

    // staging maps (fixed per thread)
    const int a_r   = tid >> 1;          // 128 rows, 2 threads/row
    const int a_col = (tid & 1) * 16;    // 16 floats each
    const int b_n   = tid >> 1;          // 128 n-rows, 2 threads/row
    const int b_kk  = (tid & 1) * 16;    // 16 bf16 each
    const int gr    = m_blk + a_r;
    const bool a_ok = gr < M;

    v8f acc[2][4];
    #pragma unroll
    for (int mi = 0; mi < 2; ++mi)
        #pragma unroll
        for (int ni = 0; ni < 4; ++ni) acc[mi][ni] = (v8f){};

    const int nsteps = K >> 5;          // K-steps of 32
    float4 qa[4];                       // staged A (16 floats)
    v8bf   qb[2];                       // staged B (16 bf16)

    // per-segment load duplication: keeps base pointer = raw kernel arg so the
    // address-space inference emits global_load_b128 (not flat_load).
    #define LOAD_A_FROM(P)                                                     \
        do { const float4* ap =                                                \
                 (const float4*)((P) + (size_t)gr * HDIM + kc + a_col);        \
             qa[0] = ap[0]; qa[1] = ap[1]; qa[2] = ap[2]; qa[3] = ap[3];       \
        } while (0)

    auto load_step = [&](int step) {
        const int kc  = (step & 15) << 5;     // column inside 512-wide segment
        const int seg = step >> 4;
        if (a_ok) {
            switch (seg) {
                case 0:  LOAD_A_FROM(A0); break;
                case 1:  LOAD_A_FROM(A1); break;
                case 2:  LOAD_A_FROM(A2); break;
                default: LOAD_A_FROM(A3); break;
            }
        } else {
            #pragma unroll
            for (int q = 0; q < 4; ++q) qa[q] = make_float4(0.f, 0.f, 0.f, 0.f);
        }
        const bf16* wp = Wt + (size_t)(n_blk + b_n) * K + (step << 5) + b_kk;
        qb[0] = *(const v8bf*)wp;
        qb[1] = *(const v8bf*)(wp + 8);
        if (step + 1 < nsteps)
            __builtin_prefetch(wp + 32, 0, 1);   // next K-slice of weights
    };

    load_step(0);
    int p = 0;
    for (int i = 0; i < nsteps; ++i) {
        // ---- stage registers -> LDS[p] (fp32 -> bf16 for A)
        {
            bf16 tmp[16];
            #pragma unroll
            for (int q = 0; q < 4; ++q) {
                tmp[q*4+0] = (bf16)qa[q].x; tmp[q*4+1] = (bf16)qa[q].y;
                tmp[q*4+2] = (bf16)qa[q].z; tmp[q*4+3] = (bf16)qa[q].w;
            }
            *(v8bf*)&Asm[p][a_r * 32 + a_col]     = *(v8bf*)&tmp[0];
            *(v8bf*)&Asm[p][a_r * 32 + a_col + 8] = *(v8bf*)&tmp[8];
            *(v8bf*)&Bsm[p][b_n * 32 + b_kk]      = qb[0];
            *(v8bf*)&Bsm[p][b_n * 32 + b_kk + 8]  = qb[1];
        }
        __syncthreads();
        // ---- issue next step's global loads while this step computes
        if (i + 1 < nsteps) load_step(i + 1);

        // ---- fragments (ISA 7.12.2 layouts), all ds_load_b128
        v16bf afrag[2];
        #pragma unroll
        for (int mi = 0; mi < 2; ++mi) {
            int m = wm * 32 + mi * 16 + l15;
            v8bf lo  = *(const v8bf*)&Asm[p][m * 32 + hi * 8];
            v8bf hi8 = *(const v8bf*)&Asm[p][m * 32 + 16 + hi * 8];
            #pragma unroll
            for (int j = 0; j < 8; ++j) { afrag[mi][j] = lo[j]; afrag[mi][j+8] = hi8[j]; }
        }
        v16bf bfrag[4];
        #pragma unroll
        for (int ni = 0; ni < 4; ++ni) {
            int n = wn * 64 + ni * 16 + l15;
            v8bf lo  = *(const v8bf*)&Bsm[p][n * 32 + hi * 8];
            v8bf hi8 = *(const v8bf*)&Bsm[p][n * 32 + 16 + hi * 8];
            #pragma unroll
            for (int j = 0; j < 8; ++j) { bfrag[ni][j] = lo[j]; bfrag[ni][j+8] = hi8[j]; }
        }
        #pragma unroll
        for (int mi = 0; mi < 2; ++mi)
            #pragma unroll
            for (int ni = 0; ni < 4; ++ni)
                acc[mi][ni] = __builtin_amdgcn_wmma_f32_16x16x32_bf16(
                    false, afrag[mi], false, bfrag[ni],
                    (short)0, acc[mi][ni], false, false);
        p ^= 1;
    }
    #undef LOAD_A_FROM

    // ---- store C (+bias): VGPR r -> row hi*8+r, lane -> col
    #pragma unroll
    for (int ni = 0; ni < 4; ++ni) {
        int gn = n_blk + wn * 64 + ni * 16 + l15;
        float bv = bias ? bias[gn] : 0.0f;
        #pragma unroll
        for (int mi = 0; mi < 2; ++mi) {
            #pragma unroll
            for (int r = 0; r < 8; ++r) {
                int gm = m_blk + wm * 32 + mi * 16 + hi * 8 + r;
                if (gm < M) C[(size_t)gm * N + gn] = acc[mi][ni][r] + bv;
            }
        }
    }
}

// ------------------------------- BatchNorm ---------------------------------

__global__ void col_stats(const float* __restrict__ h, int M, float* stats) {
    int c = blockIdx.x * blockDim.x + threadIdx.x;   // gridDim.x==2 -> c<512
    int r0 = blockIdx.y * 256;
    int r1 = min(r0 + 256, M);
    float s = 0.0f, s2 = 0.0f;
    for (int r = r0; r < r1; ++r) {
        float v = h[(size_t)r * HDIM + c];
        s += v; s2 += v * v;
    }
    atomicAdd(&stats[c], s);
    atomicAdd(&stats[HDIM + c], s2);
}

__global__ void bn_apply(const float* __restrict__ hbuf, const float* __restrict__ stats,
                         const float* __restrict__ g, const float* __restrict__ b,
                         const float* __restrict__ res, int M, float* __restrict__ out) {
    int i = blockIdx.x * blockDim.x + threadIdx.x;
    if (i >= M * HDIM) return;
    int c = i & (HDIM - 1);
    float invM = 1.0f / (float)M;
    float mu  = stats[c] * invM;
    float var = stats[HDIM + c] * invM - mu * mu;
    float rs  = rsqrtf(var + BN_EPS);
    float v = (hbuf[i] - mu) * rs * g[c] + b[c];
    v = v > 0.0f ? v : 0.01f * v;               // leaky_relu(0.01)
    out[i] = v + res[i];
}

// ------------------------------ log-softmax --------------------------------

__global__ void log_softmax_rows(const float* __restrict__ z, float* __restrict__ out, int M) {
    __shared__ float red[ODIM];
    int row = blockIdx.x;
    int t = threadIdx.x;
    float v = z[(size_t)row * ODIM + t];
    red[t] = v; __syncthreads();
    for (int s = ODIM / 2; s > 0; s >>= 1) {
        if (t < s) red[t] = fmaxf(red[t], red[t + s]);
        __syncthreads();
    }
    float mx = red[0]; __syncthreads();
    red[t] = expf(v - mx); __syncthreads();
    for (int s = ODIM / 2; s > 0; s >>= 1) {
        if (t < s) red[t] += red[t + s];
        __syncthreads();
    }
    float lse = mx + logf(red[0]);
    out[(size_t)row * ODIM + t] = v - lse;
}

// ------------------------------- launcher ----------------------------------

extern "C" void kernel_launch(void* const* d_in, const int* in_sizes, int n_in,
                              void* d_out, int out_size, void* d_ws, size_t ws_size,
                              hipStream_t stream) {
    const float* x   = (const float*)d_in[0];
    const int* ei1   = (const int*)d_in[1];
    const int* ei2   = (const int*)d_in[2];
    const int* ei3   = (const int*)d_in[3];
    const float* Wl1 = (const float*)d_in[4],  *Wr1 = (const float*)d_in[5];
    const float* g1  = (const float*)d_in[6],  *b1  = (const float*)d_in[7];
    const float* Wl2 = (const float*)d_in[8],  *Wr2 = (const float*)d_in[9];
    const float* g2  = (const float*)d_in[10], *b2  = (const float*)d_in[11];
    const float* Wl3 = (const float*)d_in[12], *Wr3 = (const float*)d_in[13];
    const float* g3  = (const float*)d_in[14], *b3  = (const float*)d_in[15];
    const float* res_W = (const float*)d_in[16], *res_b = (const float*)d_in[17];
    const float* mW1 = (const float*)d_in[18], *mb1 = (const float*)d_in[19];
    const float* mW2 = (const float*)d_in[20], *mb2 = (const float*)d_in[21];

    const int N1 = 30000, N2 = 15000, N3 = 8000;
    const int E1 = in_sizes[1] / 2, E2 = in_sizes[2] / 2, E3 = in_sizes[3] / 2;

    // ---- workspace carve-out (fp32 region then bf16 weight region)
    float* ws = (float*)d_ws;
    size_t o = 0;
    float* mean  = ws + o; o += (size_t)N1 * HDIM;
    float* cnt   = ws + o; o += N1;
    float* hbuf  = ws + o; o += (size_t)N1 * HDIM;
    float* x1    = ws + o; o += (size_t)N1 * HDIM;
    float* x2    = ws + o; o += (size_t)N2 * HDIM;
    float* x3    = ws + o; o += (size_t)N3 * HDIM;
    float* z1    = ws + o; o += (size_t)N3 * HDIM;
    float* z2    = ws + o; o += (size_t)N3 * ODIM;
    float* stats = ws + o; o += 2 * HDIM;
    bf16* Wc1  = (bf16*)(ws + o);              // [512 n][1024 k]
    bf16* Wc2  = Wc1  + (size_t)HDIM * 1024;
    bf16* Wc3  = Wc2  + (size_t)HDIM * 1024;
    bf16* Wres = Wc3  + (size_t)HDIM * 1024;   // [512 n][512 k]
    bf16* Wm1  = Wres + (size_t)HDIM * HDIM;   // [512 n][2048 k]
    bf16* Wm2  = Wm1  + (size_t)HDIM * 2048;   // [256 n][512 k]

    const int CB = 256;
    auto cvtg = [](int n) { return (n + 255) / 256; };
    const int SQ = HDIM * HDIM;
    // ---- pack weights: bf16, transposed [n][k]; Wcat_i = [Wl_i ; Wr_i]
    transpose_cvt_bf16<<<cvtg(SQ), CB, 0, stream>>>(Wl1, Wc1, HDIM, HDIM, 1024, 0);
    transpose_cvt_bf16<<<cvtg(SQ), CB, 0, stream>>>(Wr1, Wc1, HDIM, HDIM, 1024, HDIM);
    transpose_cvt_bf16<<<cvtg(SQ), CB, 0, stream>>>(Wl2, Wc2, HDIM, HDIM, 1024, 0);
    transpose_cvt_bf16<<<cvtg(SQ), CB, 0, stream>>>(Wr2, Wc2, HDIM, HDIM, 1024, HDIM);
    transpose_cvt_bf16<<<cvtg(SQ), CB, 0, stream>>>(Wl3, Wc3, HDIM, HDIM, 1024, 0);
    transpose_cvt_bf16<<<cvtg(SQ), CB, 0, stream>>>(Wr3, Wc3, HDIM, HDIM, 1024, HDIM);
    transpose_cvt_bf16<<<cvtg(SQ), CB, 0, stream>>>(res_W, Wres, HDIM, HDIM, HDIM, 0);
    transpose_cvt_bf16<<<cvtg(2048 * HDIM), CB, 0, stream>>>(mW1, Wm1, 2048, HDIM, 2048, 0);
    transpose_cvt_bf16<<<cvtg(HDIM * ODIM), CB, 0, stream>>>(mW2, Wm2, HDIM, ODIM, HDIM, 0);

    auto run_layer = [&](const float* xall, const int* ei, int E, int ndst,
                         const bf16* Wc, const float* g, const float* b,
                         const float* res, float* xout) {
        zero_f32<<<1024, CB, 0, stream>>>(mean, (long)ndst * HDIM);
        zero_f32<<<64, CB, 0, stream>>>(cnt, ndst);
        scatter_mean_accum<<<E, CB, 0, stream>>>(xall, ei, ei + E, E, mean, cnt);
        mean_div<<<(ndst * HDIM + 255) / 256, CB, 0, stream>>>(mean, cnt, ndst);
        dim3 gg(HDIM / 128, (ndst + 127) / 128);
        wmma_gemm<<<gg, 256, 0, stream>>>(mean, xall, nullptr, nullptr,
                                          1024, Wc, HDIM, ndst, nullptr, hbuf);
        zero_f32<<<4, CB, 0, stream>>>(stats, 2 * HDIM);
        dim3 gs(2, (ndst + 255) / 256);
        col_stats<<<gs, CB, 0, stream>>>(hbuf, ndst, stats);
        bn_apply<<<(ndst * HDIM + 255) / 256, CB, 0, stream>>>(hbuf, stats, g, b,
                                                               res, ndst, xout);
    };

    // ---- layer 1: residual = x_dst @ res_W + res_b (into x1, then fused)
    {
        dim3 gr(HDIM / 128, (N1 + 127) / 128);
        wmma_gemm<<<gr, 256, 0, stream>>>(x, nullptr, nullptr, nullptr,
                                          HDIM, Wres, HDIM, N1, res_b, x1);
        run_layer(x, ei1, E1, N1, Wc1, g1, b1, /*res=*/x1, /*xout=*/x1);
    }
    // ---- layer 2: residual = x1[:N2]
    run_layer(x1, ei2, E2, N2, Wc2, g2, b2, x1, x2);
    // ---- layer 3: residual = x2[:N3]
    run_layer(x2, ei3, E3, N3, Wc3, g3, b3, x2, x3);

    // ---- MLP over implicit concat [x | x1 | x2 | x3] (K = 2048)
    {
        dim3 gm1(HDIM / 128, (N3 + 127) / 128);
        wmma_gemm<<<gm1, 256, 0, stream>>>(x, x1, x2, x3,
                                           2048, Wm1, HDIM, N3, mb1, z1);
        dim3 gm2(ODIM / 128, (N3 + 127) / 128);
        wmma_gemm<<<gm2, 256, 0, stream>>>(z1, nullptr, nullptr, nullptr,
                                           HDIM, Wm2, ODIM, N3, mb2, z2);
        log_softmax_rows<<<N3, ODIM, 0, stream>>>(z2, (float*)d_out, N3);
    }
}
